// SwinBlock_21363167330604
// MI455X (gfx1250) — matrix operand order repaired and checked by reading, not compiled, not run on hardware
//
#include <hip/hip_runtime.h>
#include <hip/hip_bf16.h>
#include <math.h>

typedef _Float16 f16;
typedef __attribute__((ext_vector_type(16))) _Float16 v16h;
typedef __attribute__((ext_vector_type(8)))  _Float16 v8h;
typedef __attribute__((ext_vector_type(8)))  float    v8f;

#define NTOK   100352      // B*H*W = 32*56*56
#define CCH    192
#define NHEAD  6
#define HD     32
#define NWIN   2048        // 32 * 8 * 8
#define HID    768

static __device__ __forceinline__ v8f zero_v8f() {
    v8f z;
#pragma unroll
    for (int i = 0; i < 8; ++i) z[i] = 0.0f;
    return z;
}

static __device__ __forceinline__ v16h zero_v16h() {
    v16h z;
#pragma unroll
    for (int i = 0; i < 16; ++i) z[i] = (_Float16)0;
    return z;
}

static __device__ __forceinline__ v8f wmma_f16(v16h a, v16h b, v8f c) {
    // D = A(16x32 f16) * B(32x16 f16) + C(16x16 f32)
    return __builtin_amdgcn_wmma_f32_16x16x32_f16(
        /*neg_a=*/false, a, /*neg_b=*/false, b,
        /*c_mod=*/(short)0, c, /*reuse_a=*/false, /*reuse_b=*/false);
}

// A fragment: row-major A[m][k], lda halfs. Lane l holds row m = m0 + (l&15);
// elements 0-7 at k0 + 8*hi + (0..7), elements 8-15 at k0 + 16 + 8*hi + (0..7),
// hi = l>=16.  (ISA 05_wmma 7.12.2, 16-bit A 16x32)
static __device__ __forceinline__ v16h frag_a(const f16* A, int lda, int m, int mlim,
                                              int k0, int lane) {
    union { v16h v; v8h h[2]; } u;
    if (m < mlim) {
        const f16* p = A + (long)m * lda + k0 + 8 * (lane >> 4);
        u.h[0] = *(const v8h*)(p);
        u.h[1] = *(const v8h*)(p + 16);
    } else {
        u.v = zero_v16h();
    }
    return u.v;
}

// B fragment: B^T stored row-major Bw[n][k] (== torch-style weight (out,in)).
// Lane l holds column n = n0 + (l&15); elements 0-7 at k0 + 8*hi + (0..7), 8-15 at k0 + 16 + 8*hi + (0..7).
static __device__ __forceinline__ v16h frag_b(const f16* Bw, int ldb, int n, int nlim,
                                              int k0, int lane) {
    union { v16h v; v8h h[2]; } u;
    if (n < nlim) {
        const f16* p = Bw + (long)n * ldb + k0 + 8 * (lane >> 4);
        u.h[0] = *(const v8h*)(p);
        u.h[1] = *(const v8h*)(p + 16);
    } else {
        u.v = zero_v16h();
    }
    return u.v;
}

// ---------------------------------------------------------------- weight cvt
__global__ void f32_to_f16_kernel(const float* __restrict__ in, f16* __restrict__ out, int n) {
    int i = blockIdx.x * blockDim.x + threadIdx.x;
    if (i < n) out[i] = (f16)in[i];
}

// -------------------------------------------------- LayerNorm (+shift/window)
// SHIFTED: output row r is windowed order; source = x at spatial ((hs+3)%56,(ws+3)%56)
template <bool SHIFTED>
__global__ __launch_bounds__(256) void ln_kernel(const float* __restrict__ X,
                                                 const float* __restrict__ gamma,
                                                 const float* __restrict__ beta,
                                                 f16* __restrict__ out) {
    int r = blockIdx.x * 8 + (threadIdx.x >> 5);
    int lane = threadIdx.x & 31;
    if (r >= NTOK) return;
    long src;
    if (SHIFTED) {
        int win = r / 49, tok = r - win * 49;
        int b_ = win >> 6, wi = (win >> 3) & 7, wj = win & 7;
        int ii = tok / 7, jj = tok - ii * 7;
        int hh = wi * 7 + ii + 3; if (hh >= 56) hh -= 56;
        int ww = wj * 7 + jj + 3; if (ww >= 56) ww -= 56;
        src = ((((long)b_ * 56) + hh) * 56 + ww) * CCH;
    } else {
        src = (long)r * CCH;
    }
    float vals[6], s = 0.f, sq = 0.f;
#pragma unroll
    for (int i = 0; i < 6; ++i) {
        float v = X[src + i * 32 + lane];
        vals[i] = v; s += v; sq += v * v;
    }
#pragma unroll
    for (int off = 16; off; off >>= 1) {
        s  += __shfl_xor(s,  off, 32);
        sq += __shfl_xor(sq, off, 32);
    }
    float mu = s * (1.0f / CCH);
    float var = sq * (1.0f / CCH) - mu * mu;
    float rs = rsqrtf(var + 1e-5f);
#pragma unroll
    for (int i = 0; i < 6; ++i) {
        int c = i * 32 + lane;
        out[(long)r * CCH + c] = (f16)((vals[i] - mu) * rs * gamma[c] + beta[c]);
    }
}

// ------------------------------------------------------------- WMMA GEMM
// C[M][N] = A[M][K](f16) @ Bw[N][K]^T(f16) + epilogue(MODE)
// MODE 0: qkv  (+bias, q-scale, f16 out)
// MODE 1: proj (+bias, window-reverse+roll, +shortcut(extra), f32 out)
// MODE 2: fc1  (+bias, exact GELU, f16 out)
// MODE 3: fc2  (+bias, +residual(extra), f32 out)
template <int MODE>
__global__ __launch_bounds__(256) void gemm_wmma_kernel(const f16* __restrict__ A,
                                                        const f16* __restrict__ Bw,
                                                        const float* __restrict__ bias,
                                                        const float* __restrict__ extra,
                                                        void* __restrict__ out,
                                                        int M, int N, int K) {
    int lane = threadIdx.x & 31;
    int wave = threadIdx.x >> 5;           // 8 waves: 2(M) x 4(N)
    int m0 = blockIdx.y * 64 + (wave >> 2) * 32;
    int n0 = blockIdx.x * 128 + (wave & 3) * 32;

    v8f acc[2][2];
#pragma unroll
    for (int i = 0; i < 2; ++i)
#pragma unroll
        for (int j = 0; j < 2; ++j) acc[i][j] = zero_v8f();

    int mr0 = m0 + (lane & 15), mr1 = m0 + 16 + (lane & 15);
    int nc0 = n0 + (lane & 15), nc1 = n0 + 16 + (lane & 15);

    for (int k0 = 0; k0 < K; k0 += 32) {
        // prefetch next k-step of the streaming A operand (global_prefetch_b8)
        if (k0 + 32 < K) {
            if (mr0 < M) __builtin_prefetch(A + (long)mr0 * K + k0 + 32, 0, 3);
            if (mr1 < M) __builtin_prefetch(A + (long)mr1 * K + k0 + 32, 0, 3);
        }
        v16h a0 = frag_a(A, K, mr0, M, k0, lane);
        v16h a1 = frag_a(A, K, mr1, M, k0, lane);
        v16h b0 = frag_b(Bw, K, nc0, N, k0, lane);
        v16h b1 = frag_b(Bw, K, nc1, N, k0, lane);
        acc[0][0] = wmma_f16(a0, b0, acc[0][0]);
        acc[0][1] = wmma_f16(a0, b1, acc[0][1]);
        acc[1][0] = wmma_f16(a1, b0, acc[1][0]);
        acc[1][1] = wmma_f16(a1, b1, acc[1][1]);
    }

    int colt = lane & 15, rowoff = (lane >> 4) * 8;
#pragma unroll
    for (int mt = 0; mt < 2; ++mt)
#pragma unroll
        for (int nt = 0; nt < 2; ++nt) {
#pragma unroll
            for (int r = 0; r < 8; ++r) {
                int row = m0 + mt * 16 + rowoff + r;
                int col = n0 + nt * 16 + colt;
                if (row >= M || col >= N) continue;
                float v = acc[mt][nt][r];
                if constexpr (MODE == 0) {
                    v += bias[col];
                    if (col < CCH) v *= 0.17677669529663689f;   // 1/sqrt(32)
                    ((f16*)out)[(long)row * N + col] = (f16)v;
                } else if constexpr (MODE == 1) {
                    v += bias[col];
                    int win = row / 49, tok = row - win * 49;
                    int b_ = win >> 6, wi = (win >> 3) & 7, wj = win & 7;
                    int ii = tok / 7, jj = tok - ii * 7;
                    int hh = wi * 7 + ii + 3; if (hh >= 56) hh -= 56;
                    int ww = wj * 7 + jj + 3; if (ww >= 56) ww -= 56;
                    long sidx = ((((long)b_ * 56) + hh) * 56 + ww) * CCH + col;
                    ((float*)out)[sidx] = extra[sidx] + v;      // shortcut + proj
                } else if constexpr (MODE == 2) {
                    v += bias[col];
                    float g = 0.5f * v * (1.0f + erff(v * 0.70710678118654752f));
                    ((f16*)out)[(long)row * N + col] = (f16)g;
                } else {
                    v += bias[col] + extra[(long)row * N + col];
                    ((float*)out)[(long)row * N + col] = v;
                }
            }
        }
}

// ------------------------------------------------------------ attention
// one wave per (window, head). S = q k^T (+rel bias, softmax), O = P v.
__global__ __launch_bounds__(32) void attn_kernel(const f16* __restrict__ qkv,
                                                  const float* __restrict__ table,
                                                  const int* __restrict__ relidx,
                                                  f16* __restrict__ attn_out) {
    int blk = blockIdx.x;
    int win = blk / NHEAD, head = blk - win * NHEAD;
    int lane = threadIdx.x;

    __shared__ float S[64 * 64];                 // 16 KB
    __shared__ __align__(16) f16 P[64 * 64];     //  8 KB
    __shared__ __align__(16) f16 VT[32 * 64];    //  4 KB

    const f16* qbase = qkv + (long)win * 49 * (3 * CCH) + head * HD;
    const f16* kbase = qbase + CCH;
    const f16* vbase = qbase + 2 * CCH;

    // stage V^T[d][tok] into LDS (zero-pad tok 49..63)
    for (int idx = lane; idx < 32 * 64; idx += 32) {
        int d = idx >> 6, tok = idx & 63;
        VT[d * 64 + tok] = (tok < 49) ? vbase[(long)tok * (3 * CCH) + d] : (f16)0;
    }
    __syncthreads();

    // S = q @ k^T  (49x49 padded to 64x64), K = HD = 32 -> one wmma per tile
    for (int mt = 0; mt < 4; ++mt) {
        v16h a = frag_a(qbase, 3 * CCH, mt * 16 + (lane & 15), 49, 0, lane);
#pragma unroll
        for (int nt = 0; nt < 4; ++nt) {
            v16h b = frag_b(kbase, 3 * CCH, nt * 16 + (lane & 15), 49, 0, lane);
            v8f c = wmma_f16(a, b, zero_v8f());
            int colt = lane & 15, rowoff = (lane >> 4) * 8;
#pragma unroll
            for (int r = 0; r < 8; ++r)
                S[(mt * 16 + rowoff + r) * 64 + nt * 16 + colt] = c[r];
        }
    }
    __syncthreads();

    // +bias, softmax over valid 49 keys; P zero-padded to 64x64
    for (int row = lane; row < 64; row += 32) {
        if (row < 49) {
            float mx = -3.0e38f;
            for (int j = 0; j < 49; ++j) {
                float t = S[row * 64 + j] + table[relidx[row * 49 + j] * NHEAD + head];
                S[row * 64 + j] = t;
                mx = fmaxf(mx, t);
            }
            float sum = 0.f;
            for (int j = 0; j < 49; ++j) {
                float e = expf(S[row * 64 + j] - mx);
                S[row * 64 + j] = e;
                sum += e;
            }
            float inv = 1.0f / sum;
            for (int j = 0; j < 64; ++j)
                P[row * 64 + j] = (j < 49) ? (f16)(S[row * 64 + j] * inv) : (f16)0;
        } else {
            for (int j = 0; j < 64; ++j) P[row * 64 + j] = (f16)0;
        }
    }
    __syncthreads();

    // O = P(64x64) @ V(64x32): K = 64 -> two wmma steps
    f16* obase = attn_out + (long)win * 49 * CCH + head * HD;
    for (int mt = 0; mt < 4; ++mt) {
#pragma unroll
        for (int nt = 0; nt < 2; ++nt) {
            v8f c = zero_v8f();
#pragma unroll
            for (int kt = 0; kt < 2; ++kt) {
                v16h a = frag_a(P, 64, mt * 16 + (lane & 15), 64, kt * 32, lane);
                v16h b = frag_b(VT, 64, nt * 16 + (lane & 15), 32, kt * 32, lane);
                c = wmma_f16(a, b, c);
            }
            int colt = lane & 15, rowoff = (lane >> 4) * 8;
#pragma unroll
            for (int r = 0; r < 8; ++r) {
                int row = mt * 16 + rowoff + r;
                if (row < 49)
                    obase[(long)row * CCH + nt * 16 + colt] = (f16)c[r];
            }
        }
    }
}

// ---------------------------------------------------------------- launch
extern "C" void kernel_launch(void* const* d_in, const int* in_sizes, int n_in,
                              void* d_out, int out_size, void* d_ws, size_t ws_size,
                              hipStream_t stream) {
    (void)in_sizes; (void)n_in; (void)out_size; (void)ws_size;

    const float* x        = (const float*)d_in[0];
    const float* norm1_g  = (const float*)d_in[1];
    const float* norm1_b  = (const float*)d_in[2];
    const float* qkv_w    = (const float*)d_in[3];
    const float* qkv_b    = (const float*)d_in[4];
    const float* proj_w   = (const float*)d_in[5];
    const float* proj_b   = (const float*)d_in[6];
    const float* reltab   = (const float*)d_in[7];
    const float* norm2_g  = (const float*)d_in[8];
    const float* norm2_b  = (const float*)d_in[9];
    const float* fc1_w    = (const float*)d_in[10];
    const float* fc1_b    = (const float*)d_in[11];
    const float* fc2_w    = (const float*)d_in[12];
    const float* fc2_b    = (const float*)d_in[13];
    const int*   relidx   = (const int*)d_in[14];

    char* ws = (char*)d_ws;
    // byte offsets (all 16B aligned)
    f16*   xw   = (f16*)(ws + 0);                       // 100352*192 f16   = 38,535,168 B
    f16*   qkv  = (f16*)(ws + 38535168L);               // 100352*576 f16   = 115,605,504 B
    f16*   attn = (f16*)(ws + 154140672L);              // 100352*192 f16
    float* x1   = (float*)(ws + 192675840L);            // 100352*192 f32   = 77,070,336 B
    f16*   hbuf = (f16*)(ws + 269746176L);              // 100352*768 f16   = 154,140,672 B
    f16*   w16  = (f16*)(ws + 423886848L);              // converted weights
    f16*   xm   = xw;                                   // reuse: xw dead after QKV gemm

    f16* qkv_w16  = w16;                 // 576*192
    f16* proj_w16 = w16 + 110592;        // 192*192
    f16* fc1_w16  = w16 + 147456;        // 768*192
    f16* fc2_w16  = w16 + 294912;        // 192*768

    f32_to_f16_kernel<<<(110592 + 255) / 256, 256, 0, stream>>>(qkv_w,  qkv_w16, 110592);
    f32_to_f16_kernel<<<(36864  + 255) / 256, 256, 0, stream>>>(proj_w, proj_w16, 36864);
    f32_to_f16_kernel<<<(147456 + 255) / 256, 256, 0, stream>>>(fc1_w,  fc1_w16, 147456);
    f32_to_f16_kernel<<<(147456 + 255) / 256, 256, 0, stream>>>(fc2_w,  fc2_w16, 147456);

    // LN1 + roll(-3,-3) + window partition -> f16
    ln_kernel<true><<<NTOK / 8, 256, 0, stream>>>(x, norm1_g, norm1_b, xw);

    // QKV: (100352 x 192) @ (192 x 576)
    gemm_wmma_kernel<0><<<dim3(5, NTOK / 64), 256, 0, stream>>>(
        xw, qkv_w16, qkv_b, nullptr, qkv, NTOK, 3 * CCH, CCH);

    // window attention: one wave per (window, head)
    attn_kernel<<<NWIN * NHEAD, 32, 0, stream>>>(qkv, reltab, relidx, attn);

    // proj + window-reverse + roll(+3,+3) + shortcut -> x1 (f32)
    gemm_wmma_kernel<1><<<dim3(2, NTOK / 64), 256, 0, stream>>>(
        attn, proj_w16, proj_b, x, x1, NTOK, CCH, CCH);

    // LN2 -> f16
    ln_kernel<false><<<NTOK / 8, 256, 0, stream>>>(x1, norm2_g, norm2_b, xm);

    // FC1 + GELU: (100352 x 192) @ (192 x 768)
    gemm_wmma_kernel<2><<<dim3(6, NTOK / 64), 256, 0, stream>>>(
        xm, fc1_w16, fc1_b, nullptr, hbuf, NTOK, HID, CCH);

    // FC2 + residual: (100352 x 768) @ (768 x 192) -> d_out (f32)
    gemm_wmma_kernel<3><<<dim3(2, NTOK / 64), 256, 0, stream>>>(
        hbuf, fc2_w16, fc2_b, x1, d_out, NTOK, CCH, HID);
}
